// DynamicRouterMoE_20383914787207
// MI455X (gfx1250) — matrix operand order repaired and checked by reading, not compiled, not run on hardware
//
#include <hip/hip_runtime.h>
#include <math.h>

// ---------------- problem constants (match setup_inputs) ----------------
#define NTOK 8192      // B*T = 4*2048
#define CDIM 1024
#define EDIM 16
#define HDIM 4096
#define ROWS 64        // tokens per block tile (4 WMMA M-subtiles)
#define MT   4         // ROWS/16
#define HSLC 1024      // H columns per block (K-split for GEMM2)
#define NSLC (HDIM / HSLC)

typedef __attribute__((ext_vector_type(16))) __bf16    v16bf;
typedef __attribute__((ext_vector_type(8)))  __bf16    v8bf;
typedef __attribute__((ext_vector_type(2)))  __bf16    v2bf;
typedef __attribute__((ext_vector_type(8)))  float     v8f;
typedef __attribute__((ext_vector_type(8)))  unsigned  v8u;

// ---- pack two fp32 into one dword of two bf16 (one VALU op if possible) ----
__device__ __forceinline__ unsigned pk_bf16(float lo, float hi) {
#if __has_builtin(__builtin_amdgcn_cvt_pk_bf16_f32)
    v2bf q = __builtin_amdgcn_cvt_pk_bf16_f32(lo, hi);          // v_cvt_pk_bf16_f32
    return __builtin_bit_cast(unsigned, q);
#elif __has_builtin(__builtin_amdgcn_perm)
    // v_perm_b32: dword = {hi[31:16], lo[31:16]}  (truncation-mode bf16)
    return __builtin_amdgcn_perm(__float_as_uint(hi), __float_as_uint(lo),
                                 0x07060302u);
#else
    unsigned ul = __float_as_uint(lo), uh = __float_as_uint(hi);
    return (ul >> 16) | (uh & 0xFFFF0000u);
#endif
}

// ---- scalar fp32 -> bf16 (truncation: single shift) ----
__device__ __forceinline__ __bf16 f2bf(float f) {
    unsigned short h = (unsigned short)(__float_as_uint(f) >> 16);
    union { unsigned short s; __bf16 b; } o; o.s = h; return o.b;
}

// ---- A-fragment (16x32 bf16, M x K) from LDS, row stride `ldk` elements ----
// Wave32 layout: lanes 0-15 -> M=lane, K = kk+{0..15};
// lanes 16-31 -> M=lane-16, K = kk+{16..31}. Two ds_load_b128.
__device__ __forceinline__ v16bf load_a_lds(const __bf16* tile, int ldk, int kk, int lane) {
    const int m  = lane & 15;
    const int kb = (lane < 16) ? 0 : 16;
    const __bf16* p = tile + (size_t)m * ldk + (kk + kb);
    v8bf lo = *(const v8bf*)p;          // K = kk+kb    .. +7
    v8bf hi = *(const v8bf*)(p + 8);    // K = kk+kb+8  .. +15
    return __builtin_shufflevector(lo, hi, 0, 1, 2, 3, 4, 5, 6, 7,
                                           8, 9, 10, 11, 12, 13, 14, 15);
}

// ---- B-fragment (32x16 bf16, K x N) from row-major fp32 global ----
// Lane n = lane&15, K-half by lane half; element i <-> K = khalf + i.
// 16 coalesced b32 loads + 8 pack ops per fragment; fragment assembled as
// dwords via bit_cast (no sub-word moves).
template <int LDB>
__device__ __forceinline__ v16bf load_b_glb(const float* __restrict__ wp,
                                            int kk, int ncol, int lane) {
    const int khalf = (lane < 16) ? 0 : 16;
    const float* p = wp + (size_t)(kk + khalf) * LDB + ncol;
    v8u u;
#pragma unroll
    for (int i = 0; i < 8; ++i) {
        const float f0 = p[(size_t)(2 * i)     * LDB];   // K = khalf + 2i
        const float f1 = p[(size_t)(2 * i + 1) * LDB];   // K = khalf + 2i+1
        u[i] = pk_bf16(f0, f1);
    }
    return __builtin_bit_cast(v16bf, u);
}

// ---------------- kernel 0: zero output + expert counters ----------------
__global__ void moe_zero(float* __restrict__ out, int* __restrict__ cnt) {
    size_t i = (size_t)blockIdx.x * blockDim.x + threadIdx.x;
    size_t n = (size_t)NTOK * CDIM;
    for (; i < n; i += (size_t)gridDim.x * blockDim.x) out[i] = 0.0f;
    if (blockIdx.x == 0 && threadIdx.x < EDIM) cnt[threadIdx.x] = 0;
}

// ---------------- kernel 1: router (one wave32 per token) ----------------
__global__ __launch_bounds__(256)
void moe_router(const float* __restrict__ x, const float* __restrict__ wr,
                int* __restrict__ cnt, int* __restrict__ tokIdx, float* __restrict__ tokW) {
    const int lane = threadIdx.x & 31;
    const int tok  = blockIdx.x * 8 + (threadIdx.x >> 5);
    if (tok >= NTOK) return;

    float acc[EDIM];
#pragma unroll
    for (int e = 0; e < EDIM; ++e) acc[e] = 0.0f;

    const float* xrow = x + (size_t)tok * CDIM;
    for (int c = lane; c < CDIM; c += 32) {
        const float xv = xrow[c];
        const float* w = wr + (size_t)c * EDIM;
#pragma unroll
        for (int e = 0; e < EDIM; ++e) acc[e] += xv * w[e];
    }
#pragma unroll
    for (int e = 0; e < EDIM; ++e) {
        float v = acc[e];
#pragma unroll
        for (int off = 16; off > 0; off >>= 1) v += __shfl_xor(v, off, 32);
        acc[e] = v;
    }
    if (lane == 0) {
        int i0 = 0; float l0 = acc[0];
#pragma unroll
        for (int e = 1; e < EDIM; ++e) if (acc[e] > l0) { l0 = acc[e]; i0 = e; }
        int i1 = (i0 == 0) ? 1 : 0; float l1 = acc[i1];
#pragma unroll
        for (int e = 0; e < EDIM; ++e) if (e != i0 && acc[e] > l1) { l1 = acc[e]; i1 = e; }
        const float w0 = 1.0f / (1.0f + __expf(l1 - l0));   // softmax over top-2
        const float w1 = 1.0f - w0;
        int s0 = atomicAdd(&cnt[i0], 1);
        tokIdx[i0 * NTOK + s0] = tok; tokW[i0 * NTOK + s0] = w0;
        int s1 = atomicAdd(&cnt[i1], 1);
        tokIdx[i1 * NTOK + s1] = tok; tokW[i1 * NTOK + s1] = w1;
    }
}

// ---------------- kernel 2: fused expert FFN (bf16 WMMA, M- and K-blocked) ---
// grid: (NTOK/ROWS token tiles, NSLC H-slices, EDIM experts), 256 thr = 8 waves
// LDS: aTile ROWSx1024 bf16 (128KB) + hTile ROWSx1024 bf16 (128KB) + 64 idx/wt
#define SMEM_A  (ROWS * CDIM * 2)
#define SMEM_H  (ROWS * HSLC * 2)
#define SMEM_SZ (SMEM_A + SMEM_H + ROWS * 4 + ROWS * 4)

__global__ __launch_bounds__(256)
void moe_expert(const float* __restrict__ x,  const float* __restrict__ w1,
                const float* __restrict__ b1, const float* __restrict__ w2,
                const float* __restrict__ b2, const int* __restrict__ cnt,
                const int* __restrict__ tokIdx, const float* __restrict__ tokW,
                float* __restrict__ out) {
    extern __shared__ char smem[];
    __bf16* aTile = (__bf16*)smem;                       // [ROWS][CDIM]
    __bf16* hTile = (__bf16*)(smem + SMEM_A);            // [ROWS][HSLC]
    int*    tokS  = (int*)(smem + SMEM_A + SMEM_H);
    float*  wtS   = (float*)(tokS + ROWS);

    const int e    = blockIdx.z;
    const int hs   = blockIdx.y;                         // H slice index
    const int cntE = cnt[e];
    const int row0 = blockIdx.x * ROWS;
    if (row0 >= cntE) return;               // block-uniform: EXEC stays all-ones

    const int tid  = threadIdx.x;
    const int lane = tid & 31;
    const int wv   = tid >> 5;

    if (tid < ROWS) {
        int r = row0 + tid; int tk = 0; float wt = 0.0f;
        if (r < cntE) { tk = tokIdx[e * NTOK + r]; wt = tokW[e * NTOK + r]; }
        tokS[tid] = tk; wtS[tid] = wt;
    }
    __syncthreads();

    // gather ROWS token rows of x -> bf16 LDS tile (coalesced along C)
#pragma unroll 4
    for (int i = 0; i < (ROWS * CDIM) / 256; ++i) {
        int pos = tid + i * 256;
        int r = pos >> 10, c = pos & (CDIM - 1);
        aTile[pos] = f2bf(x[(size_t)tokS[r] * CDIM + c]);
    }
    __syncthreads();

    const int rb = (lane < 16) ? 0 : 8;     // C/D fragment row base
    const int nl = lane & 15;               // C/D fragment column lane

    // ---- GEMM1: hTile = relu(X[64x1024] * W1[1024, slice] + b1) ----
    // wave wv owns 128 H columns of the slice; 1 B-frag feeds 4 WMMAs.
    const float* w1e = w1 + (size_t)e * CDIM * HDIM;
#pragma unroll 1
    for (int nt = 0; nt < 8; ++nt) {
        const int hloc = wv * 128 + nt * 16 + nl;        // column within slice
        const int ncol = hs * HSLC + hloc;               // global H column
        v8f acc[MT] = {};
#pragma unroll 2
        for (int kk = 0; kk < CDIM; kk += 32) {
            v16bf b = load_b_glb<HDIM>(w1e, kk, ncol, lane);
#pragma unroll
            for (int mt = 0; mt < MT; ++mt) {
                v16bf a = load_a_lds(aTile + (size_t)(mt * 16) * CDIM, CDIM, kk, lane);
                acc[mt] = __builtin_amdgcn_wmma_f32_16x16x32_bf16(
                              false, a, false, b, (short)0, acc[mt], false, false);
            }
        }
        const float b1v = b1[e * HDIM + ncol];
#pragma unroll
        for (int mt = 0; mt < MT; ++mt)
#pragma unroll
            for (int r = 0; r < 8; ++r) {
                float h = acc[mt][r] + b1v;
                hTile[(size_t)(mt * 16 + rb + r) * HSLC + hloc] =
                    f2bf(h > 0.0f ? h : 0.0f);
            }
    }
    __syncthreads();

    // ---- GEMM2 (partial K): out += wt * (hTile[64xHSLC] * W2[slice, 1024]) ----
    const float* w2s = w2 + (size_t)e * HDIM * CDIM + (size_t)hs * HSLC * CDIM;
#pragma unroll 1
    for (int nt = 0; nt < 8; ++nt) {
        const int ncol = wv * 128 + nt * 16 + nl;        // output C column
        v8f acc[MT] = {};
#pragma unroll 2
        for (int kk = 0; kk < HSLC; kk += 32) {
            v16bf b = load_b_glb<CDIM>(w2s, kk, ncol, lane);
#pragma unroll
            for (int mt = 0; mt < MT; ++mt) {
                v16bf a = load_a_lds(hTile + (size_t)(mt * 16) * HSLC, HSLC, kk, lane);
                acc[mt] = __builtin_amdgcn_wmma_f32_16x16x32_bf16(
                              false, a, false, b, (short)0, acc[mt], false, false);
            }
        }
        const float b2v = (hs == 0) ? b2[e * CDIM + ncol] : 0.0f;  // add bias once
#pragma unroll
        for (int mt = 0; mt < MT; ++mt)
#pragma unroll
            for (int r = 0; r < 8; ++r) {
                const int row = mt * 16 + rb + r;
                const float y = (acc[mt][r] + b2v) * wtS[row];
                atomicAdd(&out[(size_t)tokS[row] * CDIM + ncol], y); // global_atomic_add_f32
            }
    }
}

// ---------------- host-side launcher ----------------
extern "C" void kernel_launch(void* const* d_in, const int* in_sizes, int n_in,
                              void* d_out, int out_size, void* d_ws, size_t ws_size,
                              hipStream_t stream) {
    const float* x  = (const float*)d_in[0];
    const float* wr = (const float*)d_in[1];
    const float* w1 = (const float*)d_in[2];
    const float* b1 = (const float*)d_in[3];
    const float* w2 = (const float*)d_in[4];
    const float* b2 = (const float*)d_in[5];
    float* out = (float*)d_out;

    // workspace layout: [cnt: 16 ints, padded][tokIdx: E*N ints][tokW: E*N floats]
    int*   cnt    = (int*)d_ws;
    int*   tokIdx = (int*)((char*)d_ws + 256);
    float* tokW   = (float*)((char*)d_ws + 256 + (size_t)EDIM * NTOK * 4);

    static bool attr_set = false;   // idempotent, deterministic
    if (!attr_set) {
        hipFuncSetAttribute((const void*)moe_expert,
                            hipFuncAttributeMaxDynamicSharedMemorySize, SMEM_SZ);
        attr_set = true;
    }

    moe_zero<<<2048, 256, 0, stream>>>(out, cnt);
    moe_router<<<NTOK / 8, 256, 0, stream>>>(x, wr, cnt, tokIdx, tokW);
    dim3 grid(NTOK / ROWS, NSLC, EDIM);
    moe_expert<<<grid, 256, SMEM_SZ, stream>>>(x, w1, b1, w2, b2,
                                               cnt, tokIdx, tokW, out);
}
